// Position_based_content_attention_88510686036264
// MI455X (gfx1250) — matrix operand order repaired and checked by reading, not compiled, not run on hardware
//
#include <hip/hip_runtime.h>

// ---------- problem constants (from reference) ----------
#define B_   64
#define TE_  512
#define TD_  128
#define D_   512      // 2*HE
#define NE_  384      // only first 384 columns of U survive the mask
#define KT_  16       // K tiles of 32 (512/32)
#define ET_  24       // e tiles of 16 (384/16)
#define TT_  32       // t tiles of 16 (512/16)
#define EP_  12       // e tile-pairs (ET_/2)
#define TP_  16       // t tile-pairs (TT_/2)

typedef __attribute__((ext_vector_type(16))) __bf16 v16bf;
typedef __attribute__((ext_vector_type(8)))  float  v8f;

union AFrag { uint4 q[2]; v16bf v; };

__device__ __forceinline__ unsigned short f2bf(float f) {
    unsigned int u = __float_as_uint(f);
    unsigned int r = u + 0x7FFFu + ((u >> 16) & 1u);   // round-to-nearest-even
    return (unsigned short)(r >> 16);
}

// ---------- kernel 0: phi_c[t,d] = phi_W[d, i+TE-t] + phi_b[d] ----------
__global__ __launch_bounds__(256) void phi_kernel(
    const float* __restrict__ phi_W, const float* __restrict__ phi_b,
    const int* __restrict__ ip, float* __restrict__ phi_c)
{
    int gid = blockIdx.x * 256 + threadIdx.x;          // TE_*D_ threads
    int d = gid & (D_ - 1);
    int t = gid >> 9;
    int col = ip[0] + TE_ - t;                          // in [i+1, i+TE]
    phi_c[gid] = phi_W[d * (TE_ + TD_) + col] + phi_b[d];
}

// ---------- kernel 1: had[b,t,d] = bf16(phi_c[t,d] * LSTM[b,t,d]) ----------
__global__ __launch_bounds__(256) void had_kernel(
    const float* __restrict__ LSTM, const float* __restrict__ phi_c,
    unsigned short* __restrict__ had)
{
    size_t gid = (size_t)blockIdx.x * 256 + threadIdx.x;   // B_*TE_*D_
    int d = (int)(gid & (D_ - 1));
    int t = (int)((gid >> 9) & (TE_ - 1));
    had[gid] = f2bf(LSTM[gid] * phi_c[t * D_ + d]);
}

// ---------- kernel 2: pack Ua_W^T into 32x16 bf16 B-fragments ----------
// Fragment (kt, et): lane L holds column n = et*16+(L&15), K = kt*32+16*(L>>4)+j,
// j = 0..15 stored contiguously -> contiguous K order, same as the A loads in gemm_kernel.
__global__ __launch_bounds__(256) void packB_kernel(
    const float* __restrict__ Ua_W, unsigned short* __restrict__ Bpack)
{
    int gt = blockIdx.x * 256 + threadIdx.x;           // KT_*ET_*32 = 12288
    if (gt >= KT_ * ET_ * 32) return;
    int lane = gt & 31;
    int frag = gt >> 5;
    int et = frag % ET_;
    int kt = frag / ET_;
    int n  = et * 16 + (lane & 15);                    // e index (< 384)
    int kb = kt * 32 + 16 * (lane >> 4);               // k (=d) base
    unsigned short* dst = Bpack + (size_t)frag * 512 + lane * 16;
    #pragma unroll
    for (int j = 0; j < 16; ++j)
        dst[j] = f2bf(Ua_W[n * D_ + kb + j]);          // B[k,n] = Ua_W[n,k]
}

// ---------- kernel 3: base[b] = va_b + sum_k tanh(s_i.Wa_W[k]+Wa_b[k])*va_W[k] ----------
__global__ __launch_bounds__(128) void base_kernel(
    const float* __restrict__ s_i, const float* __restrict__ Wa_W,
    const float* __restrict__ Wa_b, const float* __restrict__ va_W,
    const float* __restrict__ va_b, float* __restrict__ basep)
{
    int b = blockIdx.x, k = threadIdx.x;               // 128 threads
    const float* srow = s_i + (size_t)b * 512;
    const float* wrow = Wa_W + (size_t)k * 512;
    float dot = Wa_b[k];
    for (int j = 0; j < 512; ++j) dot += srow[j] * wrow[j];
    __shared__ float red[128];
    red[k] = tanhf(dot) * va_W[k];
    __syncthreads();
    for (int s = 64; s > 0; s >>= 1) {
        if (k < s) red[k] += red[k + s];
        __syncthreads();
    }
    if (k == 0) basep[b] = red[0] + va_b[0];
}

// ---------- kernel 4: WMMA GEMM U = had @ Ua_W^T, 2x2 register-blocked ----------
// Each wave computes a 32x32 tile of U (2 t-tiles x 2 e-tiles): per K-step,
// 2 A-frags + 2 B-frags feed 4 WMMAs (halves fragment traffic per FLOP).
// Fused epilogue: tanh(U + Ua_b)*va_W, half-wave reduce over N, write partials.
__global__ __launch_bounds__(256) void gemm_kernel(
    const unsigned short* __restrict__ had,     // [B*TE, 512] bf16 row-major
    const unsigned short* __restrict__ Bpack,   // [KT_*ET_, 512] bf16 fragments
    const float* __restrict__ Ua_b,
    const float* __restrict__ va_W,
    float* __restrict__ P)
{
    const int lane = threadIdx.x & 31;
    const int wid  = threadIdx.x >> 5;
    const int gw   = blockIdx.x * 8 + wid;             // 64*16*12 = 12288 waves
    const int ep = gw % EP_;                           // e tile-pair
    const int tp = (gw / EP_) % TP_;                   // t tile-pair
    const int b  = gw / (EP_ * TP_);

    const int half = lane >> 4;
    const int l15  = lane & 15;
    // A row pointers for the two t-tiles of this wave
    const unsigned short* arow0 = had + ((size_t)(b * TE_ + tp * 32 + l15)) * D_;
    const unsigned short* arow1 = arow0 + (size_t)16 * D_;
    // B fragment pointers for the two e-tiles of this wave
    const unsigned short* bb0 = Bpack + (size_t)(2 * ep) * 512 + lane * 16;
    const unsigned short* bb1 = bb0 + 512;

    v8f c00 = {}, c01 = {}, c10 = {}, c11 = {};
    for (int kt = 0; kt < KT_; ++kt) {
        const int k0 = kt * 32;
        if (kt + 4 < KT_) {                            // run-ahead prefetch, WGP scope
            __builtin_prefetch(arow0 + k0 + 128, 0, 3);
            __builtin_prefetch(arow1 + k0 + 128, 0, 3);
        }
        // 16-bit A 16x32 layout (contiguous, matches Bpack): half 0 -> K 0..15; half 1 -> 16..31
        AFrag a0, a1, f0, f1;
        a0.q[0] = *(const uint4*)(arow0 + k0 + 16 * half);
        a0.q[1] = *(const uint4*)(arow0 + k0 + 16 * half + 8);
        a1.q[0] = *(const uint4*)(arow1 + k0 + 16 * half);
        a1.q[1] = *(const uint4*)(arow1 + k0 + 16 * half + 8);
        const unsigned short* bp0 = bb0 + (size_t)kt * (ET_ * 512);
        const unsigned short* bp1 = bb1 + (size_t)kt * (ET_ * 512);
        f0.q[0] = *(const uint4*)(bp0);
        f0.q[1] = *(const uint4*)(bp0 + 8);
        f1.q[0] = *(const uint4*)(bp1);
        f1.q[1] = *(const uint4*)(bp1 + 8);
        c00 = __builtin_amdgcn_wmma_f32_16x16x32_bf16(false, a0.v, false, f0.v, (short)0, c00, false, false);
        c01 = __builtin_amdgcn_wmma_f32_16x16x32_bf16(false, a0.v, false, f1.v, (short)0, c01, false, false);
        c10 = __builtin_amdgcn_wmma_f32_16x16x32_bf16(false, a1.v, false, f0.v, (short)0, c10, false, false);
        c11 = __builtin_amdgcn_wmma_f32_16x16x32_bf16(false, a1.v, false, f1.v, (short)0, c11, false, false);
    }

    // Epilogue: val = tanh(U + Ua_b[n]) * va_W[128+n]; reduce over the 16 N lanes.
    v8f cc[2][2] = { { c00, c01 }, { c10, c11 } };
    #pragma unroll
    for (int j = 0; j < 2; ++j) {
        const int et = 2 * ep + j;
        const int n  = et * 16 + l15;
        const float uab = Ua_b[n];
        const float vw  = va_W[TD_ + n];
        #pragma unroll
        for (int i = 0; i < 2; ++i) {
            #pragma unroll
            for (int r = 0; r < 8; ++r) {
                float val = tanhf(cc[i][j][r] + uab) * vw;
                #pragma unroll
                for (int off = 8; off > 0; off >>= 1)
                    val += __shfl_xor(val, off, 16);   // reduce within half-wave
                if (l15 == 0) {
                    int m = tp * 32 + i * 16 + r + 8 * half;   // C layout: M = r + 8*half
                    P[((size_t)(b * TE_ + m)) * ET_ + et] = val;
                }
            }
        }
    }
}

// ---------- kernel 5: E[b,t] = base[b] + sum_et P (fixed order -> deterministic) ----------
__global__ __launch_bounds__(256) void reduceE_kernel(
    const float* __restrict__ P, const float* __restrict__ basep,
    float* __restrict__ E)
{
    int gid = blockIdx.x * 256 + threadIdx.x;          // B_*TE_
    float acc = basep[gid >> 9];
    const float* p = P + (size_t)gid * ET_;
    #pragma unroll
    for (int j = 0; j < ET_; ++j) acc += p[j];
    E[gid] = acc;
}

// ---------- kernel 6: softmax over t, then c[b,d] = sum_t a[t]*LSTM[b,t,d] ----------
__global__ __launch_bounds__(512) void softmax_ctx_kernel(
    const float* __restrict__ E, const float* __restrict__ LSTM,
    float* __restrict__ out)
{
    int b = blockIdx.x, tid = threadIdx.x;             // 512 threads
    __shared__ float sA[512];
    __shared__ float sR[512];
    float e = E[b * TE_ + tid];
    sR[tid] = e;
    __syncthreads();
    for (int s = 256; s > 0; s >>= 1) {
        if (tid < s) sR[tid] = fmaxf(sR[tid], sR[tid + s]);
        __syncthreads();
    }
    float mx = sR[0];
    __syncthreads();
    float ex = expf(e - mx);
    sR[tid] = ex;
    __syncthreads();
    for (int s = 256; s > 0; s >>= 1) {
        if (tid < s) sR[tid] += sR[tid + s];
        __syncthreads();
    }
    sA[tid] = ex / sR[0];
    __syncthreads();
    const float* base = LSTM + (size_t)b * TE_ * D_ + tid;   // d = tid
    float acc = 0.f;
    for (int t = 0; t < TE_; ++t)
        acc += sA[t] * base[(size_t)t * D_];
    out[b * D_ + tid] = acc;
}

// ---------- launcher ----------
extern "C" void kernel_launch(void* const* d_in, const int* in_sizes, int n_in,
                              void* d_out, int out_size, void* d_ws, size_t ws_size,
                              hipStream_t stream) {
    const float* LSTM  = (const float*)d_in[0];
    const float* s_i   = (const float*)d_in[1];
    const float* Wa_W  = (const float*)d_in[2];
    const float* Wa_b  = (const float*)d_in[3];
    const float* Ua_W  = (const float*)d_in[4];
    const float* Ua_b  = (const float*)d_in[5];
    const float* va_W  = (const float*)d_in[6];
    const float* va_b  = (const float*)d_in[7];
    const float* phi_W = (const float*)d_in[8];
    const float* phi_b = (const float*)d_in[9];
    const int*   ip    = (const int*)d_in[10];
    float* out = (float*)d_out;

    // workspace layout (bytes)
    char* ws = (char*)d_ws;
    unsigned short* had   = (unsigned short*)(ws + 0);          // 33,554,432
    unsigned short* Bpack = (unsigned short*)(ws + 33554432);   //    393,216
    float* E     = (float*)(ws + 33947648);                     //    131,072
    float* phi_c = (float*)(ws + 34078720);                     //  1,048,576
    float* basep = (float*)(ws + 35127296);                     //        256
    float* P     = (float*)(ws + 35127552);                     //  3,145,728  (total ~38.3 MB)

    phi_kernel        <<<(TE_ * D_) / 256, 256, 0, stream>>>(phi_W, phi_b, ip, phi_c);
    had_kernel        <<<(B_ * TE_ * D_) / 256, 256, 0, stream>>>(LSTM, phi_c, had);
    packB_kernel      <<<48, 256, 0, stream>>>(Ua_W, Bpack);
    base_kernel       <<<B_, 128, 0, stream>>>(s_i, Wa_W, Wa_b, va_W, va_b, basep);
    gemm_kernel       <<<(B_ * TP_ * EP_) / 8, 256, 0, stream>>>(had, Bpack, Ua_b, va_W, P);
    reduceE_kernel    <<<(B_ * TE_) / 256, 256, 0, stream>>>(P, basep, E);
    softmax_ctx_kernel<<<B_, 512, 0, stream>>>(E, LSTM, out);
}
